// Seq2SeqChatbot_33268816674956
// MI455X (gfx1250) — compile-verified, matrix operand order checked by reading, not run on hardware
//
#include <hip/hip_runtime.h>
#include <math.h>

// ---------------------------------------------------------------------------
// Seq2Seq chatbot forward pass for gfx1250 (MI455X), wave32 + WMMA f16.
// V=32000, E=256, H=512, B=32, S=64, T=48.
// All GEMMs are x @ W.T with W stored (N,K): ideal for v_wmma_f32_16x16x32_f16
// with both operands loaded K-major per lane (ISA 7.12.2 layout).
// GEMM is register-blocked 1x4 in N: each wave computes a 16x64 strip so the
// A fragment is reused across 4 WMMAs. Loads are issued in a batch ahead of
// the 4 WMMAs so the compiler can use partial s_wait_loadcnt overlap.
// ---------------------------------------------------------------------------

typedef __attribute__((ext_vector_type(16))) _Float16 v16h;
typedef __attribute__((ext_vector_type(8)))  _Float16 v8h;
typedef __attribute__((ext_vector_type(8)))  float    v8f;

#define VSZ 32000
#define ESZ 256
#define HSZ 512
#define BSZ 32
#define SSZ 64
#define TSZ 48

__device__ __forceinline__ float sigmoidf_dev(float x) {
    return 1.0f / (1.0f + __expf(-x));
}

// ---------------------------------------------------------------------------
// Generic WMMA GEMM: C[M,N] = A[M,K] * B[N,K]^T (+bias[n]) (optional tanh)
// (optional f16 mirror of C). One 16(M)x64(N) strip per wave, f32 accum.
// Requirements: M%16==0, N%64==0, K%32==0, lda/ldb/ldc16 multiples of 8.
// ---------------------------------------------------------------------------
__global__ void wmma_gemm_kernel(const _Float16* __restrict__ A, int lda,
                                 const _Float16* __restrict__ B, int ldb,
                                 const float* __restrict__ bias,
                                 float* __restrict__ C, long long ldc,
                                 _Float16* __restrict__ C16, int ldc16,
                                 int M, int N, int K, int act) {
    const int lane = threadIdx.x & 31;
    const int wave = threadIdx.x >> 5;
    const int wpb  = blockDim.x >> 5;
    const int tile = blockIdx.x * wpb + wave;
    const int tilesN = N >> 6;                    // 64-wide N strips
    const int tilesTotal = (M >> 4) * tilesN;
    if (tile >= tilesTotal) return;   // wave-uniform branch: EXEC stays all-1s

    const int tm = tile / tilesN;
    const int tn = tile % tilesN;

    // ISA A-matrix 16xK layout: lane L<16 holds row L, K={k0..k0+7, k0+16..k0+23};
    // lane L>=16 holds row L-16, K={k0+8..k0+15, k0+24..k0+31}.  B mirrored (N rows).
    const int rowA = tm * 16 + (lane & 15);
    const int rowB = tn * 64 + (lane & 15);
    const int koff = (lane & 16) ? 8 : 0;
    const _Float16* ap = A + (size_t)rowA * lda + koff;
    const _Float16* bp = B + (size_t)rowB * ldb + koff;
    const size_t bstep = (size_t)16 * ldb;        // next 16-row N sub-tile

    v8f acc[4];
#pragma unroll
    for (int j = 0; j < 4; ++j) acc[j] = (v8f){0.f,0.f,0.f,0.f,0.f,0.f,0.f,0.f};

    for (int k0 = 0; k0 < K; k0 += 32) {
        // Phase 1: issue ALL loads for this k-step (A + 4 B fragments).
        // In-order VMEM completion lets the compiler wait partially per WMMA.
        v8h alo = *(const v8h*)(ap + k0);
        v8h ahi = *(const v8h*)(ap + k0 + 16);
        v8h blo[4], bhi[4];
#pragma unroll
        for (int j = 0; j < 4; ++j) {
            const _Float16* bpj = bp + (size_t)j * bstep + k0;
            blo[j] = *(const v8h*)bpj;
            bhi[j] = *(const v8h*)(bpj + 16);
        }
        v16h av = __builtin_shufflevector(alo, ahi, 0,1,2,3,4,5,6,7,8,9,10,11,12,13,14,15);
        // Phase 2: 4 WMMAs reusing the A fragment.
#pragma unroll
        for (int j = 0; j < 4; ++j) {
            v16h bv = __builtin_shufflevector(blo[j], bhi[j],
                                              0,1,2,3,4,5,6,7,8,9,10,11,12,13,14,15);
            acc[j] = __builtin_amdgcn_wmma_f32_16x16x32_f16(
                /*neg_a=*/false, av, /*neg_b=*/false, bv,
                /*c_mod=*/(short)0, acc[j], /*reuse_a=*/false, /*reuse_b=*/false);
        }
    }

    // C/D layout: VGPR r -> (lanes 0-15: M=r, lanes 16-31: M=r+8), N=lane&15.
    const int m0 = tm * 16 + ((lane >> 4) << 3);
#pragma unroll
    for (int j = 0; j < 4; ++j) {
        const int n = tn * 64 + j * 16 + (lane & 15);
        const float bn = bias ? bias[n] : 0.f;
#pragma unroll
        for (int r = 0; r < 8; ++r) {
            float v = acc[j][r] + bn;
            if (act == 1) v = tanhf(v);
            const size_t m = (size_t)(m0 + r);
            C[m * (size_t)ldc + n] = v;
            if (C16) C16[m * (size_t)ldc16 + n] = (_Float16)v;
        }
    }
}

// ---------------------------------------------------------------------------
// f32 -> f16 conversion with optional row-wise concatenation of two sources:
// dst[r][0:w0] = s0[r], dst[r][w0:w0+w1] = s1[r].  (s1==null for plain copy)
// ---------------------------------------------------------------------------
__global__ void convert_cat_kernel(const float* __restrict__ s0, int w0,
                                   const float* __restrict__ s1, int w1,
                                   _Float16* __restrict__ dst, long long total) {
    long long i = (long long)blockIdx.x * 256 + threadIdx.x;
    if (i >= total) return;
    const int W = w0 + w1;
    long long r = i / W;
    int c = (int)(i % W);
    float v = (c < w0) ? s0[r * (long long)w0 + c]
                       : s1[r * (long long)w1 + (c - w0)];
    dst[i] = (_Float16)v;
}

// Encoder embedding gather -> X16[s*B+b][e], token 0 row forced to zero.
__global__ void embed_enc_kernel(const int* __restrict__ input_seq,
                                 const float* __restrict__ emb,
                                 _Float16* __restrict__ X16) {
    int i = blockIdx.x * 256 + threadIdx.x;      // < S*B*E = 2048*256
    int row = i >> 8, e = i & 255;
    int s = row >> 5, b = row & 31;
    int tok = input_seq[b * SSZ + s];
    float v = (tok == 0) ? 0.f : emb[(size_t)tok * ESZ + e];
    X16[i] = (_Float16)v;
}

__global__ void zero_f32_kernel(float* p, long long n) {
    long long i = (long long)blockIdx.x * 256 + threadIdx.x;
    if (i < n) p[i] = 0.f;
}
__global__ void zero_f16_kernel(_Float16* p, long long n) {
    long long i = (long long)blockIdx.x * 256 + threadIdx.x;
    if (i < n) p[i] = (_Float16)0.f;
}

// out[:, 0, :] = 0
__global__ void zero_t0_kernel(float* __restrict__ out) {
    int i = blockIdx.x * 256 + threadIdx.x;      // < B*V
    int b = i / VSZ, v = i % VSZ;
    out[(size_t)b * TSZ * VSZ + v] = 0.f;
}

// ---------------------------------------------------------------------------
// Encoder LSTM gate update for both directions at scan step t.
//   z = Zpre[t] + zh + bih + bhh ; c' = sig(f)c + sig(i)tanh(g); h = sig(o)tanh(c')
// Writes c, h16 (next recurrent A operand), enc_out (f32), hcat16 (f16).
// ---------------------------------------------------------------------------
__global__ void enc_gate_kernel(const float* __restrict__ zpre_f,
                                const float* __restrict__ zpre_b,
                                const float* __restrict__ zh,      // [2][32][2048]
                                const float* __restrict__ bih_f, const float* __restrict__ bhh_f,
                                const float* __restrict__ bih_b, const float* __restrict__ bhh_b,
                                float* __restrict__ cbuf,          // [2][32][512]
                                _Float16* __restrict__ henc16,     // [2][32][512]
                                float* __restrict__ enc_out,       // [32][64][1024]
                                _Float16* __restrict__ hcat16,     // [32][1024]
                                int t) {
    int i = blockIdx.x * 256 + threadIdx.x;      // < 2*32*512
    int dir = i >> 14;
    int rem = i & 16383;
    int b = rem >> 9, n = rem & 511;
    const float* zpre = dir ? zpre_b : zpre_f;
    const float* bih  = dir ? bih_b  : bih_f;
    const float* bhh  = dir ? bhh_b  : bhh_f;
    size_t zrow = ((size_t)t * BSZ + b) * 2048;
    size_t hrow = (size_t)dir * BSZ * 2048 + (size_t)b * 2048;
    float zi = zpre[zrow + n]        + zh[hrow + n]        + bih[n]        + bhh[n];
    float zf = zpre[zrow + 512 + n]  + zh[hrow + 512 + n]  + bih[512 + n]  + bhh[512 + n];
    float zg = zpre[zrow + 1024 + n] + zh[hrow + 1024 + n] + bih[1024 + n] + bhh[1024 + n];
    float zo = zpre[zrow + 1536 + n] + zh[hrow + 1536 + n] + bih[1536 + n] + bhh[1536 + n];
    float c = cbuf[i];
    c = sigmoidf_dev(zf) * c + sigmoidf_dev(zi) * tanhf(zg);
    float h = sigmoidf_dev(zo) * tanhf(c);
    cbuf[i]   = c;
    henc16[i] = (_Float16)h;
    int s = dir ? (SSZ - 1 - t) : t;
    enc_out[((size_t)b * SSZ + s) * 1024 + dir * 512 + n] = h;
    hcat16[b * 1024 + dir * 512 + n] = (_Float16)h;
}

// enc_score[b][s] = enc_out[b][s][:] . attn_w[0:1024] + attn_b
__global__ void enc_score_kernel(const float* __restrict__ enc_out,
                                 const float* __restrict__ attn_w,
                                 const float* __restrict__ attn_b,
                                 float* __restrict__ enc_score) {
    __shared__ float sdata[256];
    int row = blockIdx.x;                         // b*64 + s
    int tid = threadIdx.x;
    float p = 0.f;
    for (int d = tid; d < 1024; d += 256)
        p += enc_out[(size_t)row * 1024 + d] * attn_w[d];
    sdata[tid] = p; __syncthreads();
    for (int s = 128; s > 0; s >>= 1) {
        if (tid < s) sdata[tid] += sdata[tid + s];
        __syncthreads();
    }
    if (tid == 0) enc_score[row] = sdata[0] + attn_b[0];
}

// ---------------------------------------------------------------------------
// Decoder attention + context + embedding gather. One block per batch row.
// Writes xin16[b][0:256]=emb(f16), xin16[b][256:1280]=ctx(f16).
// ---------------------------------------------------------------------------
__global__ void attn_ctx_kernel(const float* __restrict__ hdec,      // [32][512]
                                const float* __restrict__ attn_w,    // [1536]
                                const float* __restrict__ enc_score, // [32][64]
                                const float* __restrict__ enc_out,   // [32][64][1024]
                                const float* __restrict__ emb_dec,   // [V][256]
                                const int* __restrict__ target_seq,  // [32][48]
                                int t,
                                _Float16* __restrict__ xin16) {      // [32][1792]
    __shared__ float sdata[256];
    __shared__ float wgt[64];
    __shared__ float a_sh, max_sh, inv_sh;
    const int b = blockIdx.x, tid = threadIdx.x;
    const float* wa_h = attn_w + 1024;

    float p = 0.f;
    for (int i = tid; i < 512; i += 256) p += hdec[b * 512 + i] * wa_h[i];
    sdata[tid] = p; __syncthreads();
    for (int s = 128; s > 0; s >>= 1) {
        if (tid < s) sdata[tid] += sdata[tid + s];
        __syncthreads();
    }
    if (tid == 0) a_sh = sdata[0];
    __syncthreads();

    if (tid < 64) wgt[tid] = enc_score[b * 64 + tid] + a_sh;
    __syncthreads();
    if (tid == 0) {
        float m = -INFINITY;
        for (int s = 0; s < 64; ++s) m = fmaxf(m, wgt[s]);
        max_sh = m;
    }
    __syncthreads();
    if (tid < 64) wgt[tid] = __expf(wgt[tid] - max_sh);
    __syncthreads();
    if (tid == 0) {
        float s = 0.f;
        for (int i = 0; i < 64; ++i) s += wgt[i];
        inv_sh = 1.f / s;
    }
    __syncthreads();
    if (tid < 64) wgt[tid] *= inv_sh;
    __syncthreads();

    for (int d = tid; d < 1024; d += 256) {
        float acc = 0.f;
#pragma unroll 4
        for (int s = 0; s < 64; ++s)
            acc += wgt[s] * enc_out[((size_t)b * 64 + s) * 1024 + d];
        xin16[(size_t)b * 1792 + 256 + d] = (_Float16)acc;
    }
    int tok = target_seq[b * TSZ + t];
    for (int e = tid; e < 256; e += 256) {
        float v = (tok == 0) ? 0.f : emb_dec[(size_t)tok * ESZ + e];
        xin16[(size_t)b * 1792 + e] = (_Float16)v;
    }
}

// Decoder gate: note reference passes h as BOTH h and c into lstm_cell.
__global__ void dec_gate_kernel(const float* __restrict__ z,    // [32][2048]
                                const float* __restrict__ bih,
                                const float* __restrict__ bhh,
                                float* __restrict__ hdec,        // in: h(prev)=c, out: h2
                                _Float16* __restrict__ xin16) {  // writes [b][1280:1792]
    int i = blockIdx.x * 256 + threadIdx.x;      // < 32*512
    int b = i >> 9, n = i & 511;
    size_t zr = (size_t)b * 2048;
    float zi = z[zr + n]        + bih[n]        + bhh[n];
    float zf = z[zr + 512 + n]  + bih[512 + n]  + bhh[512 + n];
    float zg = z[zr + 1024 + n] + bih[1024 + n] + bhh[1024 + n];
    float zo = z[zr + 1536 + n] + bih[1536 + n] + bhh[1536 + n];
    float c = hdec[i];                            // c := h(prev)
    c = sigmoidf_dev(zf) * c + sigmoidf_dev(zi) * tanhf(zg);
    float h = sigmoidf_dev(zo) * tanhf(c);
    hdec[i] = h;
    xin16[(size_t)b * 1792 + 1280 + n] = (_Float16)h;
}

// ---------------------------------------------------------------------------
extern "C" void kernel_launch(void* const* d_in, const int* in_sizes, int n_in,
                              void* d_out, int out_size, void* d_ws, size_t ws_size,
                              hipStream_t stream) {
    (void)in_sizes; (void)n_in; (void)out_size; (void)ws_size;

    const int*   input_seq  = (const int*)  d_in[0];
    const int*   target_seq = (const int*)  d_in[1];
    const float* emb_enc    = (const float*)d_in[2];
    const float* wih_f      = (const float*)d_in[3];
    const float* whh_f      = (const float*)d_in[4];
    const float* bih_f      = (const float*)d_in[5];
    const float* bhh_f      = (const float*)d_in[6];
    const float* wih_b      = (const float*)d_in[7];
    const float* whh_b      = (const float*)d_in[8];
    const float* bih_b      = (const float*)d_in[9];
    const float* bhh_b      = (const float*)d_in[10];
    const float* enc_fc_w   = (const float*)d_in[11];
    const float* enc_fc_b   = (const float*)d_in[12];
    const float* emb_dec    = (const float*)d_in[13];
    const float* wih_d      = (const float*)d_in[14];
    const float* whh_d      = (const float*)d_in[15];
    const float* bih_d      = (const float*)d_in[16];
    const float* bhh_d      = (const float*)d_in[17];
    const float* attn_w     = (const float*)d_in[18];
    const float* attn_b     = (const float*)d_in[19];
    const float* fc_w       = (const float*)d_in[20];
    const float* fc_b       = (const float*)d_in[21];
    float* out = (float*)d_out;

    // ---- workspace carve-out (256B aligned) ----
    char* ws = (char*)d_ws;
    size_t off = 0;
    auto carve = [&](size_t bytes) -> void* {
        off = (off + 255) & ~(size_t)255;
        void* p = ws + off;
        off += bytes;
        return p;
    };
    _Float16* X16     = (_Float16*)carve(2048ULL * 256 * 2);   // embedded inputs
    _Float16* wihf16  = (_Float16*)carve(2048ULL * 256 * 2);
    _Float16* wihb16  = (_Float16*)carve(2048ULL * 256 * 2);
    _Float16* whhf16  = (_Float16*)carve(2048ULL * 512 * 2);
    _Float16* whhb16  = (_Float16*)carve(2048ULL * 512 * 2);
    _Float16* encfc16 = (_Float16*)carve(512ULL * 1024 * 2);
    _Float16* dcat16  = (_Float16*)carve(2048ULL * 1792 * 2);  // [wih_d | whh_d]
    _Float16* fcw16   = (_Float16*)carve((size_t)VSZ * 512 * 2);
    float*    ZpreF   = (float*)carve(64ULL * 32 * 2048 * 4);  // X @ wih_f.T
    float*    ZpreB   = (float*)carve(64ULL * 32 * 2048 * 4);
    float*    zh      = (float*)carve(2ULL * 32 * 2048 * 4);
    float*    zdec    = (float*)carve(32ULL * 2048 * 4);
    float*    cbuf    = (float*)carve(2ULL * 32 * 512 * 4);
    _Float16* henc16  = (_Float16*)carve(2ULL * 32 * 512 * 2);
    float*    enc_out = (float*)carve(32ULL * 64 * 1024 * 4);
    _Float16* hcat16  = (_Float16*)carve(32ULL * 1024 * 2);
    float*    hdec    = (float*)carve(32ULL * 512 * 4);
    _Float16* xin16   = (_Float16*)carve(32ULL * 1792 * 2);
    float*    escore  = (float*)carve(32ULL * 64 * 4);

    auto nblk = [](long long n) { return (int)((n + 255) / 256); };
    auto gemm = [&](const _Float16* A, int lda, const _Float16* B, int ldb,
                    const float* bias, float* C, long long ldc,
                    _Float16* C16, int ldc16, int M, int N, int K, int act) {
        int tiles = (M / 16) * (N / 64);          // 16x64 strip per wave
        int blocks = (tiles + 7) / 8;             // 8 waves (256 threads) per block
        wmma_gemm_kernel<<<blocks, 256, 0, stream>>>(A, lda, B, ldb, bias, C, ldc,
                                                     C16, ldc16, M, N, K, act);
    };

    // ---- one-time prep ----
    zero_f16_kernel<<<nblk(2 * 32 * 512), 256, 0, stream>>>(henc16, 2LL * 32 * 512);
    zero_f32_kernel<<<nblk(2 * 32 * 512), 256, 0, stream>>>(cbuf, 2LL * 32 * 512);

    convert_cat_kernel<<<nblk(2048LL * 256), 256, 0, stream>>>(wih_f, 256, nullptr, 0, wihf16, 2048LL * 256);
    convert_cat_kernel<<<nblk(2048LL * 256), 256, 0, stream>>>(wih_b, 256, nullptr, 0, wihb16, 2048LL * 256);
    convert_cat_kernel<<<nblk(2048LL * 512), 256, 0, stream>>>(whh_f, 512, nullptr, 0, whhf16, 2048LL * 512);
    convert_cat_kernel<<<nblk(2048LL * 512), 256, 0, stream>>>(whh_b, 512, nullptr, 0, whhb16, 2048LL * 512);
    convert_cat_kernel<<<nblk(512LL * 1024), 256, 0, stream>>>(enc_fc_w, 1024, nullptr, 0, encfc16, 512LL * 1024);
    convert_cat_kernel<<<nblk(2048LL * 1792), 256, 0, stream>>>(wih_d, 1280, whh_d, 512, dcat16, 2048LL * 1792);
    convert_cat_kernel<<<nblk((long long)VSZ * 512), 256, 0, stream>>>(fc_w, 512, nullptr, 0, fcw16, (long long)VSZ * 512);

    embed_enc_kernel<<<2048, 256, 0, stream>>>(input_seq, emb_enc, X16);

    // ---- encoder: hoisted input GEMMs (one dense 2048x2048x256 WMMA GEMM each)
    gemm(X16, 256, wihf16, 256, nullptr, ZpreF, 2048, nullptr, 0, 2048, 2048, 256, 0);
    gemm(X16, 256, wihb16, 256, nullptr, ZpreB, 2048, nullptr, 0, 2048, 2048, 256, 0);

    // ---- encoder recurrent scan (both directions per step) ----
    for (int t = 0; t < SSZ; ++t) {
        gemm(henc16,             512, whhf16, 512, nullptr, zh,              2048, nullptr, 0, 32, 2048, 512, 0);
        gemm(henc16 + 32 * 512,  512, whhb16, 512, nullptr, zh + 32 * 2048,  2048, nullptr, 0, 32, 2048, 512, 0);
        enc_gate_kernel<<<128, 256, 0, stream>>>(ZpreF, ZpreB, zh,
                                                 bih_f, bhh_f, bih_b, bhh_b,
                                                 cbuf, henc16, enc_out, hcat16, t);
    }

    // ---- decoder init: h = tanh([hT_f|hT_b] @ enc_fc_w.T + b), mirrored to xin16[:,1280:]
    gemm(hcat16, 1024, encfc16, 1024, enc_fc_b, hdec, 512, xin16 + 1280, 1792, 32, 512, 1024, /*tanh*/1);

    // enc_score = enc_out . wa_e + attn_b
    enc_score_kernel<<<32 * 64, 256, 0, stream>>>(enc_out, attn_w, attn_b, escore);

    // out[:, 0, :] = 0
    zero_t0_kernel<<<nblk((long long)BSZ * VSZ), 256, 0, stream>>>(out);

    // ---- decoder scan ----
    for (int t = 0; t < TSZ - 1; ++t) {
        attn_ctx_kernel<<<32, 256, 0, stream>>>(hdec, attn_w, escore, enc_out,
                                                emb_dec, target_seq, t, xin16);
        // z = [emb|ctx|h] @ [wih_d|whh_d].T   (fused K=1792 WMMA GEMM)
        gemm(xin16, 1792, dcat16, 1792, nullptr, zdec, 2048, nullptr, 0, 32, 2048, 1792, 0);
        dec_gate_kernel<<<64, 256, 0, stream>>>(zdec, bih_d, bhh_d, hdec, xin16);
        // pred = h2 @ fc_w.T + fc_b  -> out[:, t+1, :]  (dominant GEMM, 32x32000x512)
        gemm(xin16 + 1280, 1792, fcw16, 512, fc_b,
             out + (size_t)(t + 1) * VSZ, (long long)TSZ * VSZ,
             nullptr, 0, 32, VSZ, 512, 0);
    }
}
